// EMAVectorQuantizer_6897717478034
// MI455X (gfx1250) — compile-verified, matrix operand order checked by reading, not run on hardware
//
#include <hip/hip_runtime.h>
#include <hip/hip_bf16.h>

// Problem constants (match reference setup_inputs)
#define BDIM   8
#define DDIM   256
#define TDIM   1024
#define BT     8192      // B*T
#define NCODE  16384
#define NTILES (NCODE / 16)
#define NCHUNK 8
#define TPC    (NTILES / NCHUNK)   // 128 tiles (2048 codes) per N-chunk
#define DECAYF 0.99f
#define ONEMD  0.01f
#define BETAF  0.25f
#define EPSF   1e-5f

// Padded LDS row: 256 bf16 (512 B = 128 dwords) + 8 bf16 (16 B = 4 dwords)
#define BROW   264

typedef __attribute__((ext_vector_type(16))) __bf16 v16bf;
typedef __attribute__((ext_vector_type(8)))  float  v8f;
typedef unsigned int u32x4 __attribute__((ext_vector_type(4)));
typedef int          i32x4 __attribute__((ext_vector_type(4)));
typedef int          i32x8 __attribute__((ext_vector_type(8)));

union FragU { uint4 u[2]; v16bf v; };

// ---------------------------------------------------------------------------
// TDM: DMA one 16x256 bf16 codebook tile (8 KB) global -> LDS, with LDS row
// padding of 4 dwords every 128 dwords (bank-conflict-free fragment reads).
// D# per CDNA5 ISA ch.8: group0 = {count, lds_addr, global_addr, type=2},
// group1 = {data_size=2B, pad_en, pad_interval=128dw, pad_amount=4dw,
//           tensor_dim0=256, tensor_dim1=16384, tile=[256,16], stride0=256}.
// ---------------------------------------------------------------------------
__device__ __forceinline__ void tdm_load_tile(const __hip_bfloat16* gptr,
                                              unsigned lds_addr) {
  const unsigned long long ga = (unsigned long long)(uintptr_t)gptr;
  u32x4 g0;
  g0[0] = 1u;                                                 // count=1
  g0[1] = lds_addr;                                           // LDS byte addr
  g0[2] = (unsigned)(ga & 0xFFFFFFFFu);                       // gaddr[31:0]
  g0[3] = (unsigned)((ga >> 32) & 0x01FFFFFFu) | (2u << 30);  // gaddr[56:32]|type=2
  i32x8 g1;
  g1[0] = (int)0x07910000u;        // data_size=2B | pad_en | intv=128dw | amt=4dw
  g1[1] = (int)(256u << 16);       // tensor_dim0 = 256
  g1[2] = (int)(16384u << 16);     // tensor_dim1 = 16384 (lo16)
  g1[3] = (int)(256u << 16);       // tile_dim0 = 256
  g1[4] = 16;                      // tile_dim1 = 16, tile_dim2 = 0
  g1[5] = 256;                     // tensor_dim0_stride = 256
  g1[6] = 0;
  g1[7] = 0;
  i32x4 gz = {0, 0, 0, 0};
#if defined(__clang_major__) && (__clang_major__ >= 23)
  i32x8 gz8 = {0, 0, 0, 0, 0, 0, 0, 0};
  __builtin_amdgcn_tensor_load_to_lds(g0, g1, gz, gz, gz8, 0);
#else
  __builtin_amdgcn_tensor_load_to_lds(g0, g1, gz, gz, 0);
#endif
}

// ---------------------------------------------------------------------------
// K1: weight [N,D] f32 -> bf16 copy + exact f32 row norms ||w||^2
// ---------------------------------------------------------------------------
__global__ __launch_bounds__(256) void vq_prep_weight(
    const float* __restrict__ w, __hip_bfloat16* __restrict__ wb,
    float* __restrict__ wnorm) {
  const int n = blockIdx.x;
  const int d = threadIdx.x;
  const float v = w[(size_t)n * DDIM + d];
  wb[(size_t)n * DDIM + d] = __float2bfloat16(v);
  __shared__ float red[256];
  red[d] = v * v;
  __syncthreads();
  for (int s = 128; s > 0; s >>= 1) {
    if (d < s) red[d] += red[d + s];
    __syncthreads();
  }
  if (d == 0) wnorm[n] = red[0];
}

// ---------------------------------------------------------------------------
// K2: z [B,D,T] f32 -> zf [B*T, D] bf16 (LDS-tiled transpose per batch)
// ---------------------------------------------------------------------------
__global__ __launch_bounds__(256) void vq_transpose_z(
    const float* __restrict__ z, __hip_bfloat16* __restrict__ zb) {
  __shared__ float tile[32][33];
  const int b  = blockIdx.z;
  const int t0 = blockIdx.x * 32;
  const int d0 = blockIdx.y * 32;
  const int tx = threadIdx.x;
  const int ty = threadIdx.y;
  #pragma unroll
  for (int j = 0; j < 4; ++j) {
    int row = ty + j * 8;
    tile[row][tx] = z[((size_t)b * DDIM + (d0 + row)) * TDIM + (t0 + tx)];
  }
  __syncthreads();
  #pragma unroll
  for (int j = 0; j < 4; ++j) {
    int row = ty + j * 8;
    zb[((size_t)b * TDIM + (t0 + row)) * DDIM + (d0 + tx)] =
        __float2bfloat16(tile[tx][row]);
  }
}

// ---------------------------------------------------------------------------
// K3: fused bf16-WMMA GEMM (zf @ w^T) + running argmin over a 2048-code chunk.
// Grid = (BT/256 M-blocks, NCHUNK N-chunks). Each of the 8 waves owns a 32-row
// M-tile (two A fragment sets in registers, so every LDS B fragment feeds two
// WMMAs -> half the LDS bytes/FLOP, 4 independent accumulator chains). All
// waves share a TDM-staged double-buffered 16x256 codebook tile in LDS.
// dist = ||w_n||^2 - 2*<z_m,w_n>  (||z||^2 constant per row, dropped)
// ---------------------------------------------------------------------------
__global__ __launch_bounds__(256) void vq_argmin(
    const __hip_bfloat16* __restrict__ zb,   // [BT][D]
    const __hip_bfloat16* __restrict__ wb,   // [N][D]
    const float* __restrict__ wnorm,         // [N]
    float* __restrict__ pval,                // [NCHUNK][BT] partial min dist
    int*   __restrict__ pidx) {              // [NCHUNK][BT] partial argmin
  __shared__ __align__(16) __hip_bfloat16 btile[2][16][BROW];

  const int wave = threadIdx.x >> 5;   // 0..7
  const int lane = threadIdx.x & 31;
  const int lr   = lane & 15;
  const int hi   = lane >> 4;
  const int m0   = blockIdx.x * 256 + wave * 32;
  const int ntBase = blockIdx.y * TPC;

  // Two A fragment sets (rows m0..m0+15 and m0+16..m0+31), full K=256.
  FragU a0[8], a1[8];
  const __hip_bfloat16* zrow0 = zb + (size_t)(m0 + lr) * DDIM;
  const __hip_bfloat16* zrow1 = zb + (size_t)(m0 + 16 + lr) * DDIM;
  #pragma unroll
  for (int kk = 0; kk < 8; ++kk) {
    const int k0 = kk * 32;
    a0[kk].u[0] = *(const uint4*)(zrow0 + k0 + hi * 8);
    a0[kk].u[1] = *(const uint4*)(zrow0 + k0 + 16 + hi * 8);
    a1[kk].u[0] = *(const uint4*)(zrow1 + k0 + hi * 8);
    a1[kk].u[1] = *(const uint4*)(zrow1 + k0 + 16 + hi * 8);
  }

  float minv0[8], minv1[8];
  int   mini0[8], mini1[8];
  #pragma unroll
  for (int j = 0; j < 8; ++j) {
    minv0[j] = 3.4e38f; mini0[j] = 0;
    minv1[j] = 3.4e38f; mini1[j] = 0;
  }

  // Prologue: wave 0 kicks off the TDM for the chunk's first tile.
  if (threadIdx.x < 32) {
    tdm_load_tile(wb + (size_t)ntBase * 16 * DDIM,
                  (unsigned)(uintptr_t)&btile[0][0][0]);
  }

  for (int it = 0; it < TPC; ++it) {
    const int cur = it & 1;
    if (threadIdx.x < 32) {
      if (it + 1 < TPC) {
        tdm_load_tile(wb + (size_t)(ntBase + it + 1) * 16 * DDIM,
                      (unsigned)(uintptr_t)&btile[cur ^ 1][0][0]);
        __builtin_amdgcn_s_wait_tensorcnt(1);   // tile `it` landed in LDS
      } else {
        __builtin_amdgcn_s_wait_tensorcnt(0);
      }
    }
    __syncthreads();   // btile[cur] visible to all waves

    const __hip_bfloat16* brow = &btile[cur][lr][0];
    v8f ce0 = {}, co0 = {}, ce1 = {}, co1 = {};
    #pragma unroll
    for (int kk = 0; kk < 8; kk += 2) {
      const int k0 = kk * 32;
      FragU b0, b1;
      b0.u[0] = *(const uint4*)(brow + k0 + hi * 8);
      b0.u[1] = *(const uint4*)(brow + k0 + 16 + hi * 8);
      b1.u[0] = *(const uint4*)(brow + k0 + 32 + hi * 8);
      b1.u[1] = *(const uint4*)(brow + k0 + 48 + hi * 8);
      ce0 = __builtin_amdgcn_wmma_f32_16x16x32_bf16(
                false, a0[kk].v, false, b0.v, (short)0, ce0, false, false);
      ce1 = __builtin_amdgcn_wmma_f32_16x16x32_bf16(
                false, a1[kk].v, false, b0.v, (short)0, ce1, false, false);
      co0 = __builtin_amdgcn_wmma_f32_16x16x32_bf16(
                false, a0[kk + 1].v, false, b1.v, (short)0, co0, false, false);
      co1 = __builtin_amdgcn_wmma_f32_16x16x32_bf16(
                false, a1[kk + 1].v, false, b1.v, (short)0, co1, false, false);
    }

    // C layout: lane holds column n; VGPR j -> row m = j + 8*hi.
    const int   n  = (ntBase + it) * 16 + lr;
    const float wn = wnorm[n];
    #pragma unroll
    for (int j = 0; j < 8; ++j) {
      const float d0 = wn - 2.0f * (ce0[j] + co0[j]);
      const float d1 = wn - 2.0f * (ce1[j] + co1[j]);
      if (d0 < minv0[j]) { minv0[j] = d0; mini0[j] = n; }  // strict < keeps first n
      if (d1 < minv1[j]) { minv1[j] = d1; mini1[j] = n; }
    }
    __syncthreads();   // all waves done reading btile[cur] before reuse
  }

  // Reduce over the 16 lanes of each half (the n direction of the tile).
  #pragma unroll
  for (int j = 0; j < 8; ++j) {
    float v0 = minv0[j]; int i0 = mini0[j];
    float v1 = minv1[j]; int i1 = mini1[j];
    #pragma unroll
    for (int off = 1; off < 16; off <<= 1) {
      const float ov0 = __shfl_xor(v0, off, 32);
      const int   oi0 = __shfl_xor(i0, off, 32);
      const float ov1 = __shfl_xor(v1, off, 32);
      const int   oi1 = __shfl_xor(i1, off, 32);
      if (ov0 < v0 || (ov0 == v0 && oi0 < i0)) { v0 = ov0; i0 = oi0; }
      if (ov1 < v1 || (ov1 == v1 && oi1 < i1)) { v1 = ov1; i1 = oi1; }
    }
    minv0[j] = v0; mini0[j] = i0;
    minv1[j] = v1; mini1[j] = i1;
  }

  // Lanes 0 / 16 hold final per-chunk results for this wave's 32 rows.
  if (lr == 0) {
    const size_t base = (size_t)blockIdx.y * BT;
    #pragma unroll
    for (int j = 0; j < 8; ++j) {
      const int mA = m0 + j + 8 * hi;
      const int mB = m0 + 16 + j + 8 * hi;
      pval[base + mA] = minv0[j];  pidx[base + mA] = mini0[j];
      pval[base + mB] = minv1[j];  pidx[base + mB] = mini1[j];
    }
  }
}

// ---------------------------------------------------------------------------
// K3b: combine per-chunk partials (ascending n order -> first-min semantics),
// write final idx (int + float) and code-usage counts.
// ---------------------------------------------------------------------------
__global__ __launch_bounds__(256) void vq_argmin_combine(
    const float* __restrict__ pval, const int* __restrict__ pidx,
    int* __restrict__ idx_i32, float* __restrict__ idx_f,
    float* __restrict__ counts) {
  const int m = blockIdx.x * 256 + threadIdx.x;
  float bv = pval[m];
  int   bi = pidx[m];
  #pragma unroll
  for (int c = 1; c < NCHUNK; ++c) {
    const float v = pval[(size_t)c * BT + m];
    const int  ii = pidx[(size_t)c * BT + m];
    if (v < bv) { bv = v; bi = ii; }
  }
  idx_i32[m] = bi;
  idx_f[m]   = (float)bi;
  atomicAdd(&counts[bi], 1.0f);
}

// ---------------------------------------------------------------------------
// K4: gather z_q -> out[B,D,T], commitment-loss partials, scatter embed_sum.
// Linear index i == ((b*D+d)*T + t) matches z layout exactly.
// ---------------------------------------------------------------------------
__global__ __launch_bounds__(256) void vq_gather_out(
    const float* __restrict__ z, const float* __restrict__ weight,
    const int* __restrict__ idx, float* __restrict__ out,
    float* __restrict__ embed_sum, float* __restrict__ loss_acc) {
  const size_t i = (size_t)blockIdx.x * 256 + threadIdx.x;
  const int t  = (int)(i % TDIM);
  const int bd = (int)(i / TDIM);
  const int d  = bd % DDIM;
  const int b  = bd / DDIM;
  const int bt = b * TDIM + t;
  const int n  = idx[bt];
  const float wv = weight[(size_t)n * DDIM + d];
  const float zv = z[i];
  out[i] = wv;
  atomicAdd(&embed_sum[(size_t)n * DDIM + d], zv);
  const float diff = wv - zv;
  __shared__ float red[256];
  red[threadIdx.x] = diff * diff;
  __syncthreads();
  for (int s = 128; s > 0; s >>= 1) {
    if ((int)threadIdx.x < s) red[threadIdx.x] += red[threadIdx.x + s];
    __syncthreads();
  }
  if (threadIdx.x == 0) atomicAdd(loss_acc, red[0]);
}

// ---------------------------------------------------------------------------
// K5: new_cluster_size = decay*cs + (1-decay)*counts; accumulate total.
// ---------------------------------------------------------------------------
__global__ __launch_bounds__(256) void vq_cluster(
    const float* __restrict__ cs, const float* __restrict__ counts,
    float* __restrict__ out_ncs, float* __restrict__ ntotal) {
  const int n = blockIdx.x * 256 + threadIdx.x;
  const float nc = DECAYF * cs[n] + ONEMD * counts[n];
  out_ncs[n] = nc;
  __shared__ float red[256];
  red[threadIdx.x] = nc;
  __syncthreads();
  for (int s = 128; s > 0; s >>= 1) {
    if ((int)threadIdx.x < s) red[threadIdx.x] += red[threadIdx.x + s];
    __syncthreads();
  }
  if (threadIdx.x == 0) atomicAdd(ntotal, red[0]);
}

// ---------------------------------------------------------------------------
// K6: new_embed_avg + Laplace-smoothed new_weight.
// ---------------------------------------------------------------------------
__global__ __launch_bounds__(256) void vq_ema_weight(
    const float* __restrict__ ea, const float* __restrict__ es,
    const float* __restrict__ ncs, const float* __restrict__ ntotal,
    float* __restrict__ out_nea, float* __restrict__ out_nw) {
  const size_t i = (size_t)blockIdx.x * 256 + threadIdx.x;
  const int n = (int)(i / DDIM);
  const float nea = DECAYF * ea[i] + ONEMD * es[i];
  out_nea[i] = nea;
  const float nt = *ntotal;
  const float sm = (ncs[n] + EPSF) / (nt + (float)NCODE * EPSF) * nt;
  out_nw[i] = nea / sm;
}

// ---------------------------------------------------------------------------
// K7: perplexity from counts + finalize loss scalar.
// ---------------------------------------------------------------------------
__global__ __launch_bounds__(256) void vq_finalize(
    const float* __restrict__ counts, const float* __restrict__ loss_acc,
    float* __restrict__ out_perp, float* __restrict__ out_loss) {
  float s = 0.0f;
  for (int n = threadIdx.x; n < NCODE; n += 256) {
    const float p = counts[n] * (1.0f / (float)BT);
    s += p * __logf(p + 1e-10f);
  }
  __shared__ float red[256];
  red[threadIdx.x] = s;
  __syncthreads();
  for (int st = 128; st > 0; st >>= 1) {
    if ((int)threadIdx.x < st) red[threadIdx.x] += red[threadIdx.x + st];
    __syncthreads();
  }
  if (threadIdx.x == 0) {
    *out_perp = __expf(-red[0]);
    *out_loss = BETAF * (*loss_acc) / (float)((size_t)BDIM * TDIM * DDIM);
  }
}

// ---------------------------------------------------------------------------
// Launch
// ---------------------------------------------------------------------------
extern "C" void kernel_launch(void* const* d_in, const int* in_sizes, int n_in,
                              void* d_out, int out_size, void* d_ws, size_t ws_size,
                              hipStream_t stream) {
  const float* z      = (const float*)d_in[0];   // [B,D,T]
  const float* weight = (const float*)d_in[1];   // [N,D]
  const float* cs     = (const float*)d_in[2];   // [N]
  const float* ea     = (const float*)d_in[3];   // [N,D]

  // Workspace layout (bytes)
  char* ws = (char*)d_ws;
  const size_t WS_ZB     = 0;                                   // BT*D bf16 = 4 MB
  const size_t WS_WB     = WS_ZB     + (size_t)BT * DDIM * 2;   // N*D bf16 = 8 MB
  const size_t WS_WNORM  = WS_WB     + (size_t)NCODE * DDIM * 2;
  const size_t WS_COUNTS = WS_WNORM  + (size_t)NCODE * 4;
  const size_t WS_IDX    = WS_COUNTS + (size_t)NCODE * 4;
  const size_t WS_SCAL   = WS_IDX    + (size_t)BT * 4;
  const size_t WS_ESUM   = WS_SCAL   + 256;                     // N*D f32 = 16 MB
  const size_t WS_PVAL   = WS_ESUM   + (size_t)NCODE * DDIM * 4;
  const size_t WS_PIDX   = WS_PVAL   + (size_t)NCHUNK * BT * 4;
  const size_t WS_END    = WS_PIDX   + (size_t)NCHUNK * BT * 4;

  __hip_bfloat16* zb = (__hip_bfloat16*)(ws + WS_ZB);
  __hip_bfloat16* wb = (__hip_bfloat16*)(ws + WS_WB);
  float* wnorm     = (float*)(ws + WS_WNORM);
  float* counts    = (float*)(ws + WS_COUNTS);
  int*   idx_i32   = (int*)  (ws + WS_IDX);
  float* loss_acc  = (float*)(ws + WS_SCAL);
  float* ntotal    = (float*)(ws + WS_SCAL) + 1;
  float* embed_sum = (float*)(ws + WS_ESUM);
  float* pval      = (float*)(ws + WS_PVAL);
  int*   pidx      = (int*)  (ws + WS_PIDX);

  // Output sections (concatenated flat, float)
  float* out    = (float*)d_out;
  float* o_out  = out;                        // 2097152
  float* o_perp = out + 2097152;              // 1
  float* o_idx  = out + 2097153;              // 8192
  float* o_loss = out + 2105345;              // 1
  float* o_nw   = out + 2105346;              // 4194304
  float* o_ncs  = out + 6299650;              // 16384
  float* o_nea  = out + 6316034;              // 4194304

  // Zero the accumulation regions (counts, idx, scalars, embed_sum)
  hipMemsetAsync(ws + WS_COUNTS, 0, WS_PVAL - WS_COUNTS, stream);

  vq_prep_weight<<<NCODE, 256, 0, stream>>>(weight, wb, wnorm);
  vq_transpose_z<<<dim3(TDIM / 32, DDIM / 32, BDIM), dim3(32, 8), 0, stream>>>(z, zb);
  vq_argmin<<<dim3(BT / 256, NCHUNK), 256, 0, stream>>>(zb, wb, wnorm, pval, pidx);
  vq_argmin_combine<<<BT / 256, 256, 0, stream>>>(pval, pidx, idx_i32, o_idx, counts);
  vq_gather_out<<<(BDIM * DDIM * TDIM) / 256, 256, 0, stream>>>(
      z, weight, idx_i32, o_out, embed_sum, loss_acc);
  vq_cluster<<<NCODE / 256, 256, 0, stream>>>(cs, counts, o_ncs, ntotal);
  vq_ema_weight<<<(NCODE * DDIM) / 256, 256, 0, stream>>>(
      ea, embed_sum, o_ncs, ntotal, o_nea, o_nw);
  vq_finalize<<<1, 256, 0, stream>>>(counts, loss_acc, o_perp, o_loss);
}
